// att_84104049590807
// MI455X (gfx1250) — matrix operand order repaired and checked by reading, not compiled, not run on hardware
//
#include <hip/hip_runtime.h>

// ---------- CDNA5 WMMA types / helpers ----------
typedef __attribute__((ext_vector_type(16))) __bf16 v16bf;
typedef __attribute__((ext_vector_type(8)))  float  v8f;

struct U8b { unsigned int u[8]; };
struct F8  { float f[8]; };

__device__ __forceinline__ v16bf pack_frag(uint4 lo, uint4 hi) {
  U8b t;
  t.u[0]=lo.x; t.u[1]=lo.y; t.u[2]=lo.z; t.u[3]=lo.w;
  t.u[4]=hi.x; t.u[5]=hi.y; t.u[6]=hi.z; t.u[7]=hi.w;
  return __builtin_bit_cast(v16bf, t);
}
__device__ __forceinline__ v16bf zero16() { U8b t = {}; return __builtin_bit_cast(v16bf, t); }
__device__ __forceinline__ v8f   zero8()  { F8  t = {}; return __builtin_bit_cast(v8f, t); }

__device__ __forceinline__ v8f wmma_bf16(v16bf a, v16bf b, v8f c) {
  // D = A(16x32 bf16) * B(32x16 bf16) + C(16x16 f32)
  return __builtin_amdgcn_wmma_f32_16x16x32_bf16(false, a, false, b, (short)0, c, false, false);
}

__device__ __forceinline__ unsigned short f32_to_bf16(float f) {
  unsigned int u = __builtin_bit_cast(unsigned int, f);
  u += 0x7FFFu + ((u >> 16) & 1u);   // round-to-nearest-even
  return (unsigned short)(u >> 16);
}
__device__ __forceinline__ float lrelu(float x) { return x > 0.f ? x : 0.01f * x; }

// ---------- problem constants ----------
constexpr int Bn  = 16;     // batch
constexpr int S   = 4096;   // seq
constexpr int Dd  = 512;    // hidden
constexpr int L   = 50;     // labels
constexpr int NCH = 400;    // L * 8 channels
constexpr int CH1 = 64;           // K1 s-chunk rows (64x512 bf16 = 64KB LDS)
constexpr int NCHUNK1 = S / CH1;  // 64
constexpr int CH3 = 32;           // K3 s-chunk rows
constexpr int SSPLIT = 8;         // K3 S-splits (deterministic partials, 512 blocks)

// ---------- K0: des = max_c(label); w = lrelu(des@Ww+bw); bias; label->bf16 ----------
__global__ void __launch_bounds__(256) k0_prep(
    const float* __restrict__ lab, const float* __restrict__ Ww,
    const float* __restrict__ bw,  const float* __restrict__ Wb,
    const float* __restrict__ bb,
    unsigned short* __restrict__ labbf, float* __restrict__ wbuf,
    float* __restrict__ bias) {
  __shared__ float des[Dd];
  __shared__ float red[256];
  int l = blockIdx.x, t = threadIdx.x;
  const float* labl = lab + (size_t)l * 8 * Dd;
  for (int i = t; i < 8 * Dd; i += 256)
    labbf[(size_t)l * 8 * Dd + i] = f32_to_bf16(labl[i]);
  for (int d = t; d < Dd; d += 256) {
    float m = labl[d];
    #pragma unroll
    for (int c = 1; c < 8; ++c) m = fmaxf(m, labl[c * Dd + d]);
    des[d] = m;
  }
  __syncthreads();
  for (int d = t; d < Dd; d += 256) {
    float acc = bw[d];
    for (int k = 0; k < Dd; ++k) acc = fmaf(des[k], Ww[(size_t)k * Dd + d], acc);
    wbuf[(size_t)l * Dd + d] = lrelu(acc);
  }
  float p = fmaf(des[t], Wb[t], des[t + 256] * Wb[t + 256]);
  red[t] = p; __syncthreads();
  for (int s2 = 128; s2 > 0; s2 >>= 1) { if (t < s2) red[t] += red[t + s2]; __syncthreads(); }
  if (t == 0) bias[l] = lrelu(red[0] + bb[0]);
}

// ---------- K1: scores via WMMA, per-chunk softmax partials (max, sumexp) ----------
__global__ void __launch_bounds__(256) k1_score_stats(
    const float* __restrict__ X, const unsigned short* __restrict__ labbf,
    float* __restrict__ stats) {
  __shared__ __align__(16) unsigned short ldsX[CH1 * Dd];  // 64 KB bf16 staging
  int b = blockIdx.y, chunk = blockIdx.x, t = threadIdx.x;
  const float4* src = (const float4*)(X + ((size_t)b * S + (size_t)chunk * CH1) * Dd);
  for (int i = t; i < CH1 * Dd / 4; i += 256) {
    float4 v = src[i];
    unsigned short* d = &ldsX[i * 4];
    d[0] = f32_to_bf16(v.x); d[1] = f32_to_bf16(v.y);
    d[2] = f32_to_bf16(v.z); d[3] = f32_to_bf16(v.w);
  }
  __syncthreads();
  int wv = t >> 5, lane = t & 31, h = lane >> 4, ln = lane & 15;
  for (int cg = wv; cg < NCH / 16; cg += 8) {     // wave-uniform channel groups
    int ch = cg * 16 + ln;
    const unsigned short* brow = labbf + (size_t)ch * Dd;
    v8f acc[4];
    #pragma unroll
    for (int r = 0; r < 4; ++r) acc[r] = zero8();
    for (int kk = 0; kk < Dd; kk += 32) {
      const uint4* bp = (const uint4*)(brow + kk + 8 * h);    // B: lane=col, K halves 8h / 16+8h
      v16bf bf = pack_frag(bp[0], bp[2]);
      #pragma unroll
      for (int rt = 0; rt < 4; ++rt) {                        // A: lane=row, K halves 8h / 16+8h
        const uint4* ap = (const uint4*)&ldsX[(rt * 16 + ln) * Dd + kk + 8 * h];
        v16bf af = pack_frag(ap[0], ap[2]);
        acc[rt] = wmma_bf16(af, bf, acc[rt]);
      }
    }
    float mx = -3.4e38f;
    #pragma unroll
    for (int rt = 0; rt < 4; ++rt) {
      #pragma unroll
      for (int r2 = 0; r2 < 8; ++r2) mx = fmaxf(mx, acc[rt][r2]);
    }
    mx = fmaxf(mx, __shfl_xor(mx, 16, 32));   // combine row-halves per column
    float se = 0.f;
    #pragma unroll
    for (int rt = 0; rt < 4; ++rt) {
      #pragma unroll
      for (int r2 = 0; r2 < 8; ++r2) se += __expf(acc[rt][r2] - mx);
    }
    se += __shfl_xor(se, 16, 32);
    if (lane < 16) {
      size_t o = ((size_t)(b * NCHUNK1 + chunk) * NCH + ch) * 2;
      stats[o] = mx; stats[o + 1] = se;
    }
  }
}

// ---------- K2: combine chunk partials -> global (M, Z) per (b, channel) ----------
__global__ void __launch_bounds__(256) k2_combine(
    const float* __restrict__ stats, float* __restrict__ Mz) {
  int idx = blockIdx.x * 256 + threadIdx.x;  // 16*400 = 6400 threads exactly
  int b = idx / NCH, ch = idx % NCH;
  float M = -3.4e38f;
  for (int c = 0; c < NCHUNK1; ++c)
    M = fmaxf(M, stats[((size_t)(b * NCHUNK1 + c) * NCH + ch) * 2]);
  float Z = 0.f;
  for (int c = 0; c < NCHUNK1; ++c) {
    size_t o = ((size_t)(b * NCHUNK1 + c) * NCH + ch) * 2;
    Z += stats[o + 1] * __expf(stats[o] - M);
  }
  Mz[(size_t)(b * NCH + ch) * 2] = M;
  Mz[(size_t)(b * NCH + ch) * 2 + 1] = Z;
}

// ---------- K3: rescore -> weights -> m += Wᵀ·X (both via WMMA) ----------
__global__ void __launch_bounds__(256) k3_weight_gemm(
    const float* __restrict__ X, const unsigned short* __restrict__ labbf,
    const float* __restrict__ Mz, float* __restrict__ mpart) {
  __shared__ __align__(16) unsigned short ldsX[CH3 * Dd];  // 32 KB
  __shared__ __align__(16) unsigned short ldsW[16 * CH3];  // 1 KB weights (bf16)
  int ss = blockIdx.x, lg = blockIdx.y, b = blockIdx.z;
  int t = threadIdx.x, wv = t >> 5, lane = t & 31, h = lane >> 4, ln = lane & 15;
  int ch0 = lg * 128 + wv * 16;
  int chB = ch0 + ln;
  bool valid = (chB < NCH);                 // uniform per wave (16-aligned boundary)
  float Msub = 0.f, Zinv = 1.f;
  if (valid) {
    Msub = Mz[(size_t)(b * NCH + chB) * 2];
    Zinv = 1.f / Mz[(size_t)(b * NCH + chB) * 2 + 1];
  }
  const unsigned short* brow = valid ? (labbf + (size_t)chB * Dd) : labbf;
  v8f macc[4];
  #pragma unroll
  for (int r = 0; r < 4; ++r) macc[r] = zero8();

  constexpr int NC3 = S / SSPLIT / CH3;     // 16 chunks per block
  for (int c = 0; c < NC3; ++c) {
    int row0 = ss * (S / SSPLIT) + c * CH3;
    const float4* src = (const float4*)(X + ((size_t)b * S + row0) * Dd);
    if (c + 1 < NC3)
      __builtin_prefetch(X + ((size_t)b * S + row0 + CH3) * Dd + threadIdx.x * 4, 0, 1);
    for (int i = t; i < CH3 * Dd / 4; i += 256) {
      float4 v = src[i];
      unsigned short* d = &ldsX[i * 4];
      d[0] = f32_to_bf16(v.x); d[1] = f32_to_bf16(v.y);
      d[2] = f32_to_bf16(v.z); d[3] = f32_to_bf16(v.w);
    }
    __syncthreads();
    // scores: this wave's 16 channels x 32 rows
    v8f sc[2]; sc[0] = zero8(); sc[1] = zero8();
    for (int kk = 0; kk < Dd; kk += 32) {
      v16bf bf;
      if (valid) { const uint4* bp = (const uint4*)(brow + kk + 8 * h); bf = pack_frag(bp[0], bp[2]); }
      else bf = zero16();
      #pragma unroll
      for (int rt = 0; rt < 2; ++rt) {
        const uint4* ap = (const uint4*)&ldsX[(rt * 16 + ln) * Dd + kk + 8 * h];
        v16bf af = pack_frag(ap[0], ap[2]);
        sc[rt] = wmma_bf16(af, bf, sc[rt]);
      }
    }
    // weights: w[s,l] = sum over this label's 8 channels of exp(score-M)/Z
    #pragma unroll
    for (int rt = 0; rt < 2; ++rt) {
      #pragma unroll
      for (int r2 = 0; r2 < 8; ++r2) {
        float e = __expf(sc[rt][r2] - Msub) * Zinv;
        e += __shfl_xor(e, 1, 32);
        e += __shfl_xor(e, 2, 32);
        e += __shfl_xor(e, 4, 32);          // sum over 8-channel group
        if ((lane & 7) == 0) {
          int li   = wv * 2 + ((lane >> 3) & 1);   // local label 0..15
          int srow = rt * 16 + r2 + 8 * h;         // local s 0..31
          ldsW[li * CH3 + srow] = f32_to_bf16(e);
        }
      }
    }
    __syncthreads();
    // GEMM2: m[16l x 64d-slice] += W(16x32) * X(32x16) per d-tile, K=32 in one WMMA
    {
      const uint4* wp = (const uint4*)&ldsW[ln * CH3 + 16 * h];
      v16bf af = pack_frag(wp[0], wp[1]);
      #pragma unroll
      for (int dt = 0; dt < 4; ++dt) {
        int dcol = wv * 64 + dt * 16 + ln;
        U8b u;
        #pragma unroll
        for (int e2 = 0; e2 < 8; ++e2) {
          unsigned lo = ldsX[(2 * e2     + 16 * h) * Dd + dcol];
          unsigned hi = ldsX[(2 * e2 + 1 + 16 * h) * Dd + dcol];
          u.u[e2] = lo | (hi << 16);
        }
        v16bf bf2 = __builtin_bit_cast(v16bf, u);
        macc[dt] = wmma_bf16(af, bf2, macc[dt]);
      }
    }
    __syncthreads();
  }
  // deterministic partial write: mpart[ss][b][64 labels (padded)][512]
  #pragma unroll
  for (int dt = 0; dt < 4; ++dt) {
    #pragma unroll
    for (int r2 = 0; r2 < 8; ++r2) {
      int l = lg * 16 + r2 + 8 * h;
      int d = wv * 64 + dt * 16 + ln;
      mpart[(((size_t)ss * Bn + b) * 64 + l) * Dd + d] = macc[dt][r2];
    }
  }
}

// ---------- K4: logits[b,l] = sum_d (sum_ss m) * w + bias ----------
__global__ void __launch_bounds__(256) k4_logits(
    const float* __restrict__ mpart, const float* __restrict__ wbuf,
    const float* __restrict__ bias, float* __restrict__ out) {
  __shared__ float red[256];
  int bl = blockIdx.x; int b = bl / L, l = bl % L;
  int t = threadIdx.x;
  float p = 0.f;
  for (int d = t; d < Dd; d += 256) {
    float m = 0.f;
    #pragma unroll
    for (int ssi = 0; ssi < SSPLIT; ++ssi)
      m += mpart[(((size_t)ssi * Bn + b) * 64 + l) * Dd + d];
    p = fmaf(m, wbuf[(size_t)l * Dd + d], p);
  }
  red[t] = p; __syncthreads();
  for (int s2 = 128; s2 > 0; s2 >>= 1) { if (t < s2) red[t] += red[t + s2]; __syncthreads(); }
  if (t == 0) out[b * L + l] = red[0] + bias[l];
}

// ---------- launch ----------
extern "C" void kernel_launch(void* const* d_in, const int* in_sizes, int n_in,
                              void* d_out, int out_size, void* d_ws, size_t ws_size,
                              hipStream_t stream) {
  (void)in_sizes; (void)n_in; (void)out_size; (void)ws_size;
  const float* X   = (const float*)d_in[0];
  // d_in[1] = text_mask (unused by reference)
  const float* lab = (const float*)d_in[2];
  const float* Ww  = (const float*)d_in[3];
  const float* bw  = (const float*)d_in[4];
  const float* Wb  = (const float*)d_in[5];
  const float* bb  = (const float*)d_in[6];
  float* out = (float*)d_out;

  char* ws = (char*)d_ws;                       // 16B-aligned offsets, ~20.6 MB total
  unsigned short* labbf = (unsigned short*)(ws + 0);        // 400*512 bf16 = 409600
  float* wbuf  = (float*)(ws + 409600);                     // 50*512 f32  = 102400
  float* bias  = (float*)(ws + 512000);                     // 64 f32      = 256
  float* stats = (float*)(ws + 512256);                     // 16*64*400*2 = 3276800
  float* Mz    = (float*)(ws + 3789056);                    // 16*400*2    = 51200
  float* mpart = (float*)(ws + 3840256);                    // 8*16*64*512*4 = 16777216

  k0_prep<<<L, 256, 0, stream>>>(lab, Ww, bw, Wb, bb, labbf, wbuf, bias);
  k1_score_stats<<<dim3(NCHUNK1, Bn), 256, 0, stream>>>(X, labbf, stats);
  k2_combine<<<25, 256, 0, stream>>>(stats, Mz);
  k3_weight_gemm<<<dim3(SSPLIT, 4, Bn), 256, 0, stream>>>(X, labbf, Mz, mpart);
  k4_logits<<<Bn * L, 256, 0, stream>>>(mpart, wbuf, bias, out);
}
